// GMPool_masking_RT_67181878444566
// MI455X (gfx1250) — matrix operand order repaired and checked by reading, not compiled, not run on hardware
//
#include <hip/hip_runtime.h>

// ---------------- problem constants (from reference) ----------------
#define B_       16
#define T_       176
#define R_       116
#define HID_     128
#define SURVIVE_ 88
#define KPAD_    128               // K padded to 4 chunks of 32 for WMMA f16
#define MTOT_    (B_ * T_ * T_)    // 495616 rows through the MLP

// output layout (element offsets into d_out, all float32)
#define FC_OFF   0
#define GM_OFF   (B_ * T_ * R_)              // 326656
#define SOFT_OFF (GM_OFF + B_ * T_ * T_)     // 822272

#define NWG_GEMM (B_ * T_)                   // 2816 workgroups, one per (b,i)

typedef __attribute__((ext_vector_type(16))) _Float16 v16h;
typedef __attribute__((ext_vector_type(8)))  float    v8f;

// raw v_sqrt_f32 (no IEEE refinement) -- feeds fp16 WMMA, accuracy is ample
#define FAST_SQRTF(x) __builtin_amdgcn_sqrtf(x)

// --------------------------------------------------------------------
// CDNA5 async DMA: memory -> LDS, 16B per lane, tracked by ASYNCcnt.
// GVS mode: mem_addr = SADDR(64) + VADDR(32) ; LDS addr = VDST VGPR.
// Generic shared pointers on amdgcn carry the LDS offset in their low
// 32 bits (ISA: flat->LDS uses addr[31:0]), so the truncating cast is
// the LDS byte address.
// --------------------------------------------------------------------
__device__ __forceinline__ void async_ld_b128(void* lds_ptr, const void* gbase,
                                              unsigned byte_off)
{
    const unsigned lds_addr = (unsigned)(uintptr_t)lds_ptr;
    asm volatile("global_load_async_to_lds_b128 %0, %1, %2"
                 :
                 : "v"(lds_addr), "v"(byte_off), "s"(gbase)
                 : "memory");
}

__device__ __forceinline__ void wait_async_zero()
{
    asm volatile("s_wait_asynccnt 0x0" ::: "memory");
}

// --------------------------------------------------------------------
// Shared GEMM core: one wave computes a 16(j) x 128(hidden) strip of
//   h = dist @ w1.T   via  v_wmma_f32_16x16x32_f16
// dist row j uses xb[i][:] and xb[j][:] (both staged in LDS).
// A-fragment lane mapping used here (contiguous k, not the ISA table order):
//   lane L, half h:  row = L&15,  k = 32c + (L&16 ? 16:0) + h
// B-fragment (16-bit B 32x16): lane L: col = L&15 (+nt*16), k = 32c + (L&16 ?16:0) + h
// C/D (f32 16x16): lane L, vgpr v: row = v + (L&16 ? 8:0), col = L&15
// --------------------------------------------------------------------
__device__ __forceinline__ void dist_gemm_strip(const float* __restrict__ xb,
                                                const _Float16* __restrict__ w1h,
                                                int i, int j0, int lane,
                                                v8f acc[8])
{
    const int   lrow = lane & 15;
    const int   hi   = (lane >> 4) & 1;
    const float* xi  = xb + i * R_;
    const float* xj  = xb + (j0 + lrow) * R_;

#pragma unroll
    for (int nt = 0; nt < 8; ++nt)
#pragma unroll
        for (int v = 0; v < 8; ++v) acc[nt][v] = 0.0f;

#pragma unroll
    for (int c = 0; c < 4; ++c) {
        // A fragment: dist = sqrt((xi-xj)^2 + 1e-9), zero-padded past R.
        v16h af;
#pragma unroll
        for (int h = 0; h < 16; ++h) {
            const int k = c * 32 + hi * 16 + h;
            float d = 0.0f;
            if (k < R_) {
                const float df = xi[k] - xj[k];
                d = FAST_SQRTF(df * df + 1e-9f);
            }
            af[h] = (_Float16)d;
        }
#pragma unroll
        for (int nt = 0; nt < 8; ++nt) {
            const _Float16* src = w1h + (nt * 16 + lrow) * KPAD_ + c * 32 + hi * 16;
            v16h bf;
#pragma unroll
            for (int h = 0; h < 16; ++h) bf[h] = src[h];
            acc[nt] = __builtin_amdgcn_wmma_f32_16x16x32_f16(
                false, af, false, bf, (short)0, acc[nt], false, false);
        }
    }
}

// --------------------------------------------------------------------
// Pass 1: GEMM + deterministic per-workgroup BN partial sums (sum, sum^2)
// grid = 2816 (one per (b,i)), block = 256 (8 waves, 11 strips of j)
// --------------------------------------------------------------------
__global__ __launch_bounds__(256)
void k_gemm_stats(const float* __restrict__ X, const float* __restrict__ w1,
                  const float* __restrict__ b1, float* __restrict__ part)
{
    __shared__ __align__(16) float xb[T_ * R_];   // X[b] tile: 81664 B
    __shared__ _Float16 w1h[HID_ * KPAD_];        // w1 in fp16, k-padded: 32768 B
    __shared__ float    b1s[HID_];
    __shared__ float    wsum[8 * HID_];           // per-wave partial sums
    __shared__ float    wsq[8 * HID_];

    const int tid = threadIdx.x;
    const int wg  = blockIdx.x;                   // b*T + i
    const int b   = wg / T_;
    const int i   = wg % T_;

    // async DMA the X[b] tile straight into LDS (no VGPR round trip)
    {
        const float* gb = X + (size_t)b * T_ * R_;
        for (int idx = tid; idx < (T_ * R_) / 4; idx += 256)
            async_ld_b128(&xb[idx * 4], gb, (unsigned)(idx * 16));
    }
    for (int idx = tid; idx < HID_ * KPAD_; idx += 256) {
        const int col = idx >> 7, k = idx & 127;
        w1h[idx] = (k < R_) ? (_Float16)w1[col * R_ + k] : (_Float16)0.0f;
    }
    for (int idx = tid; idx < HID_; idx += 256) b1s[idx] = b1[idx];
    for (int idx = tid; idx < 8 * HID_; idx += 256) { wsum[idx] = 0.0f; wsq[idx] = 0.0f; }
    wait_async_zero();
    __syncthreads();

    const int wave = tid >> 5;
    const int lane = tid & 31;
    const int lrow = lane & 15;
    const int hi   = (lane >> 4) & 1;

    for (int s = wave; s < 11; s += 8) {          // 11 strips cover j = 0..175
        v8f acc[8];
        dist_gemm_strip(xb, w1h, i, s * 16, lane, acc);

#pragma unroll
        for (int nt = 0; nt < 8; ++nt) {
            const int   col = nt * 16 + lrow;
            const float b1c = b1s[col];
            float sm = 0.0f, sq = 0.0f;
#pragma unroll
            for (int v = 0; v < 8; ++v) {
                const float h = acc[nt][v] + b1c;
                sm += h; sq += h * h;
            }
            // lanes L and L^16 hold the same column -> pair-combine
            sm += __shfl_xor(sm, 16, 32);
            sq += __shfl_xor(sq, 16, 32);
            if (!hi) { wsum[wave * HID_ + col] += sm; wsq[wave * HID_ + col] += sq; }
        }
    }
    __syncthreads();

    // fixed-order reduction over the 8 waves -> deterministic partials
    for (int idx = tid; idx < 2 * HID_; idx += 256) {
        const int    col = idx & 127;
        const float* src = (idx >> 7) ? wsq : wsum;
        float a = 0.0f;
        for (int w = 0; w < 8; ++w) a += src[w * HID_ + col];
        part[(size_t)wg * 256 + idx] = a;
    }
}

// --------------------------------------------------------------------
// Fold partials -> BN affine: hn = h*a[col] + bb[col]
// --------------------------------------------------------------------
__global__ __launch_bounds__(256)
void k_finalize(const float* __restrict__ part, const float* __restrict__ gamma,
                const float* __restrict__ beta, float* __restrict__ stats)
{
    __shared__ float red[256];
    const int tid = threadIdx.x;
    float acc = 0.0f;
    for (int wg = 0; wg < NWG_GEMM; ++wg) acc += part[(size_t)wg * 256 + tid];
    red[tid] = acc;
    __syncthreads();
    if (tid < HID_) {
        const float inv = 1.0f / (float)MTOT_;
        const float mu  = red[tid] * inv;
        const float var = red[HID_ + tid] * inv - mu * mu;   // biased var (torch BN fwd)
        const float a   = gamma[tid] * rsqrtf(var + 1e-5f);
        stats[tid]        = a;
        stats[HID_ + tid] = beta[tid] - mu * a;
    }
}

// --------------------------------------------------------------------
// Pass 2: GEMM + BN + ReLU + (.@ w2.T + b2) + sigmoid, diag=1 -> grouping_M
// --------------------------------------------------------------------
__global__ __launch_bounds__(256)
void k_gemm_clf(const float* __restrict__ X, const float* __restrict__ w1,
                const float* __restrict__ b1, const float* __restrict__ w2,
                const float* __restrict__ b2, const float* __restrict__ stats,
                float* __restrict__ gm)
{
    __shared__ __align__(16) float xb[T_ * R_];
    __shared__ _Float16 w1h[HID_ * KPAD_];
    __shared__ float    b1s[HID_], a_l[HID_], bb_l[HID_], w2_l[HID_];

    const int tid = threadIdx.x;
    const int wg  = blockIdx.x;
    const int b   = wg / T_;
    const int i   = wg % T_;

    {
        const float* gb = X + (size_t)b * T_ * R_;
        for (int idx = tid; idx < (T_ * R_) / 4; idx += 256)
            async_ld_b128(&xb[idx * 4], gb, (unsigned)(idx * 16));
    }
    for (int idx = tid; idx < HID_ * KPAD_; idx += 256) {
        const int col = idx >> 7, k = idx & 127;
        w1h[idx] = (k < R_) ? (_Float16)w1[col * R_ + k] : (_Float16)0.0f;
    }
    for (int idx = tid; idx < HID_; idx += 256) {
        b1s[idx]  = b1[idx];
        a_l[idx]  = stats[idx];
        bb_l[idx] = stats[HID_ + idx];
        w2_l[idx] = w2[idx];
    }
    wait_async_zero();
    __syncthreads();

    const int   wave = tid >> 5;
    const int   lane = tid & 31;
    const int   lrow = lane & 15;
    const int   hi   = (lane >> 4) & 1;
    const float b2v  = b2[0];

    for (int s = wave; s < 11; s += 8) {
        const int j0 = s * 16;
        v8f acc[8];
        dist_gemm_strip(xb, w1h, i, j0, lane, acc);

        float rp[8];
#pragma unroll
        for (int v = 0; v < 8; ++v) rp[v] = 0.0f;
#pragma unroll
        for (int nt = 0; nt < 8; ++nt) {
            const int   col = nt * 16 + lrow;
            const float b1c = b1s[col], ac = a_l[col], bc = bb_l[col], wc = w2_l[col];
#pragma unroll
            for (int v = 0; v < 8; ++v) {
                float h = (acc[nt][v] + b1c) * ac + bc;    // BatchNorm affine
                h = fmaxf(h, 0.0f);                        // ReLU
                rp[v] += wc * h;                           // second Linear (128 -> 1)
            }
        }
#pragma unroll
        for (int v = 0; v < 8; ++v) {
            float r = rp[v];
            r += __shfl_xor(r, 1, 32);
            r += __shfl_xor(r, 2, 32);
            r += __shfl_xor(r, 4, 32);
            r += __shfl_xor(r, 8, 32);     // 16-lane butterfly within each half-wave
            if (lrow == 0) {
                const int j = j0 + v + hi * 8;
                float val = 1.0f / (1.0f + __expf(-(r + b2v)));
                if (j == i) val = 1.0f;                    // diagonal -> 1
                gm[(size_t)wg * T_ + j] = val;
            }
        }
    }
}

// --------------------------------------------------------------------
// Per-batch symmetric eigensolver (parallel Jacobi, tournament ordering)
// A (124KB) and V (124KB) both live in the 320KB WGP LDS. Then: sort
// eigenvalues descending (index permutation), gumbel argmax (tau=1e-4),
// survive mask, FC_output = X * me * eig_w, soft_indices.
// grid = 16 (one per b), block = 512
// --------------------------------------------------------------------
__global__ __launch_bounds__(512)
void k_eigen(const float* __restrict__ gm, const float* __restrict__ X,
             const float* __restrict__ unif, float* __restrict__ fc,
             float* __restrict__ soft)
{
    __shared__ __align__(16) float Amat[T_ * T_];   // 123904 B
    __shared__ float Vmat[T_ * T_];                 // 123904 B
    __shared__ float csC[88], csS[88];
    __shared__ float eval[T_];
    __shared__ int   eord[T_];
    __shared__ float coef[T_];

    const int tid = threadIdx.x;
    const int b   = blockIdx.x;

    // async DMA grouping_M[b] into LDS; init V = I on the normal path
    {
        const float* gb = gm + (size_t)b * T_ * T_;
        for (int idx = tid; idx < (T_ * T_) / 4; idx += 512)
            async_ld_b128(&Amat[idx * 4], gb, (unsigned)(idx * 16));
    }
    for (int idx = tid; idx < T_ * T_; idx += 512)
        Vmat[idx] = ((idx / T_) == (idx % T_)) ? 1.0f : 0.0f;
    wait_async_zero();
    __syncthreads();

    const int NP = T_ - 1;            // 175 rounds/sweep, 88 disjoint pairs/round
    for (int sweep = 0; sweep < 10; ++sweep) {
        for (int rnd = 0; rnd < NP; ++rnd) {
            if (tid < 88) {           // rotation angles from pre-update A
                const int home = (rnd + tid) % NP;
                const int away = (tid == 0) ? NP : (NP - tid + rnd) % NP;
                const int p = min(home, away), q = max(home, away);
                const float apq = Amat[p * T_ + q];
                float c = 1.0f, s = 0.0f;
                if (fabsf(apq) > 1e-20f) {
                    const float tau = (Amat[q * T_ + q] - Amat[p * T_ + p]) / (2.0f * apq);
                    const float t   = copysignf(1.0f, tau) /
                                      (fabsf(tau) + FAST_SQRTF(1.0f + tau * tau));
                    c = rsqrtf(1.0f + t * t);
                    s = t * c;
                }
                csC[tid] = c; csS[tid] = s;
            }
            __syncthreads();
            // phase 1: rotate columns (p,q) of A and of V  (pairs disjoint)
            for (int t = tid; t < 88 * T_; t += 512) {
                const int k = t / T_, r = t % T_;
                const int home = (rnd + k) % NP;
                const int away = (k == 0) ? NP : (NP - k + rnd) % NP;
                const int p = min(home, away), q = max(home, away);
                const float c = csC[k], s = csS[k];
                const float arp = Amat[r * T_ + p], arq = Amat[r * T_ + q];
                Amat[r * T_ + p] = c * arp - s * arq;
                Amat[r * T_ + q] = s * arp + c * arq;
                const float vrp = Vmat[r * T_ + p], vrq = Vmat[r * T_ + q];
                Vmat[r * T_ + p] = c * vrp - s * vrq;
                Vmat[r * T_ + q] = s * vrp + c * vrq;
            }
            __syncthreads();
            // phase 2: rotate rows (p,q) of A
            for (int t = tid; t < 88 * T_; t += 512) {
                const int k = t / T_, cc = t % T_;
                const int home = (rnd + k) % NP;
                const int away = (k == 0) ? NP : (NP - k + rnd) % NP;
                const int p = min(home, away), q = max(home, away);
                const float c = csC[k], s = csS[k];
                const float apc = Amat[p * T_ + cc], aqc = Amat[q * T_ + cc];
                Amat[p * T_ + cc] = c * apc - s * aqc;
                Amat[q * T_ + cc] = s * apc + c * aqc;
            }
            __syncthreads();
        }
    }

    if (tid < T_) { eval[tid] = Amat[tid * T_ + tid]; eord[tid] = tid; }
    __syncthreads();

    // even-odd transposition sort, descending, with index permutation
    for (int pass = 0; pass < T_; ++pass) {
        if (tid < 88) {
            const int f = 2 * tid + (pass & 1);
            if (f + 1 < T_ && eval[f] < eval[f + 1]) {
                const float tv = eval[f]; eval[f] = eval[f + 1]; eval[f + 1] = tv;
                const int   ti = eord[f]; eord[f] = eord[f + 1]; eord[f + 1] = ti;
            }
        }
        __syncthreads();
    }

    // gumbel softmax with tau=1e-4 == argmax over (V[:,order]*5000 + g)
    if (tid < T_) {
        const float* un = unif + ((size_t)b * T_ + tid) * T_;
        float best = -3.4e38f; int bj = 0;
        for (int j = 0; j < T_; ++j) {
            const float g  = -__logf(-__logf(un[j]));
            const float sc = Vmat[tid * T_ + eord[j]] * 5000.0f + g;
            if (sc > best) { best = sc; bj = j; }
        }
        soft[(size_t)b * T_ + tid] = (float)bj;
        const float me = (bj < SURVIVE_) ? 1.0f : 0.0f;   // isin(arange(88))
        coef[tid] = me * eval[bj];                         // me * eigenvalue gather
    }
    __syncthreads();

    // FC_output[b,t,r] = X[b,t,r] * coef[t]
    for (int u = tid; u < T_ * R_; u += 512)
        fc[(size_t)b * T_ * R_ + u] = X[(size_t)b * T_ * R_ + u] * coef[u / R_];
}

// --------------------------------------------------------------------
extern "C" void kernel_launch(void* const* d_in, const int* in_sizes, int n_in,
                              void* d_out, int out_size, void* d_ws, size_t ws_size,
                              hipStream_t stream)
{
    const float* X     = (const float*)d_in[0];
    const float* unif  = (const float*)d_in[1];
    const float* w1    = (const float*)d_in[2];
    const float* b1    = (const float*)d_in[3];
    const float* gamma = (const float*)d_in[4];
    const float* beta  = (const float*)d_in[5];
    const float* w2    = (const float*)d_in[6];
    const float* b2    = (const float*)d_in[7];

    float* out  = (float*)d_out;
    float* fc   = out + FC_OFF;
    float* gm   = out + GM_OFF;
    float* soft = out + SOFT_OFF;

    float* part  = (float*)d_ws;                      // [2816][256] BN partials
    float* stats = part + (size_t)NWG_GEMM * 256;     // a[128], bb[128]

    hipLaunchKernelGGL(k_gemm_stats, dim3(NWG_GEMM), dim3(256), 0, stream, X, w1, b1, part);
    hipLaunchKernelGGL(k_finalize,   dim3(1),        dim3(256), 0, stream, part, gamma, beta, stats);
    hipLaunchKernelGGL(k_gemm_clf,   dim3(NWG_GEMM), dim3(256), 0, stream, X, w1, b1, w2, b2, stats, gm);
    hipLaunchKernelGGL(k_eigen,      dim3(B_),       dim3(512), 0, stream, gm, X, unif, fc, soft);
}
